// QURNNCell_4337916969454
// MI455X (gfx1250) — matrix operand order repaired and checked by reading, not compiled, not run on hardware
//
#include <hip/hip_runtime.h>

// ---------------------------------------------------------------------------
// QURNN cell for MI455X (gfx1250): bf16 WMMA GEMMs (f32 accumulate) + fused
// activation epilogues + elementwise glue. B=4096, D_IN=D_H=1024.
// ---------------------------------------------------------------------------

#define B_DIM   4096
#define D_H     1024
#define D_IN    1024
#define K_BIG   2048   // D_IN + D_H

typedef __bf16 v16bf __attribute__((ext_vector_type(16)));
typedef __bf16 v8bf  __attribute__((ext_vector_type(8)));
typedef float  v8f   __attribute__((ext_vector_type(8)));

__device__ __forceinline__ v16bf cat16(v8bf lo, v8bf hi) {
  return __builtin_shufflevector(lo, hi, 0,1,2,3,4,5,6,7,8,9,10,11,12,13,14,15);
}

__device__ __forceinline__ float fast_sigmoid(float x) {
  return 1.0f / (1.0f + __expf(-x));
}
__device__ __forceinline__ float fast_tanh(float x) {
  x = fminf(fmaxf(x, -15.0f), 15.0f);
  float e = __expf(2.0f * x);
  return (e - 1.0f) / (e + 1.0f);
}

// ---------------------------------------------------------------------------
// GEMM: out[m,n] = act( sum_k A[m,k]*W[n,k] + bias[n] )
// A: [M, lda] bf16 row-major, W: [N, ldw] bf16 row-major (ldw == K)
// Workgroup = 256 threads (8 waves, 2x4), WG tile 128(M) x 128(N),
// wave tile 64(M) x 32(N) -> 4x2 v_wmma_f32_16x16x32_bf16 accumulators.
// grid = (N/128, M/128). act: 0 = sigmoid, 1 = tanh.
// ---------------------------------------------------------------------------
__global__ __launch_bounds__(256)
void gemm_bf16_wmma(const __bf16* __restrict__ A, int lda,
                    const __bf16* __restrict__ W, int ldw,
                    const float* __restrict__ bias,
                    float* __restrict__ out, int ldo,
                    int K, int act)
{
  const int lane  = threadIdx.x & 31;
  const int wave  = threadIdx.x >> 5;
  const int waveM = wave >> 2;   // 0..1
  const int waveN = wave & 3;    // 0..3
  const int half  = lane >> 4;   // 0/1: K-half selector per ISA bf16 layouts
  const int r     = lane & 15;

  const int rowBase = blockIdx.y * 128 + waveM * 64;  // + mt*16
  const int colBase = blockIdx.x * 128 + waveN * 32;  // + nt*16

  v8f acc[4][2];
#pragma unroll
  for (int mt = 0; mt < 4; ++mt)
#pragma unroll
    for (int nt = 0; nt < 2; ++nt)
      acc[mt][nt] = (v8f){0.f,0.f,0.f,0.f,0.f,0.f,0.f,0.f};

  // Per-lane base pointers (A: lane holds row M = base + r; W: lane holds
  // weight row N = base + r, i.e. B-matrix column — K-contiguous in memory).
  const __bf16* Arow[4];
#pragma unroll
  for (int mt = 0; mt < 4; ++mt)
    Arow[mt] = A + (size_t)(rowBase + mt * 16 + r) * lda;
  const __bf16* Wrow[2];
#pragma unroll
  for (int nt = 0; nt < 2; ++nt)
    Wrow[nt] = W + (size_t)(colBase + nt * 16 + r) * ldw;

#pragma unroll 2
  for (int k0 = 0; k0 < K; k0 += 32) {
    v16bf afrag[4], bfrag[2];
#pragma unroll
    for (int mt = 0; mt < 4; ++mt) {
      // A 16x32 bf16 layout: lanes hold K = half*8 + [0..7] and 16+half*8+[0..7]
      v8bf lo = *(const v8bf*)(Arow[mt] + k0 + half * 8);
      v8bf hi = *(const v8bf*)(Arow[mt] + k0 + 16 + half * 8);
      afrag[mt] = cat16(lo, hi);
    }
#pragma unroll
    for (int nt = 0; nt < 2; ++nt) {
      // B 32x16 bf16 layout: lane n holds K = half*8 + [0..7] and 16+half*8+[0..7] for column n
      const v8bf* p = (const v8bf*)(Wrow[nt] + k0 + half * 8);
      bfrag[nt] = cat16(p[0], p[2]);
    }
#pragma unroll
    for (int mt = 0; mt < 4; ++mt)
#pragma unroll
      for (int nt = 0; nt < 2; ++nt)
        acc[mt][nt] = __builtin_amdgcn_wmma_f32_16x16x32_bf16(
            false, afrag[mt], false, bfrag[nt],
            (short)0, acc[mt][nt], false, false);
  }

  // Epilogue: bias + activation, store f32.
#pragma unroll
  for (int nt = 0; nt < 2; ++nt) {
    const int col = colBase + nt * 16 + r;
    const float bn = bias[col];
#pragma unroll
    for (int mt = 0; mt < 4; ++mt) {
#pragma unroll
      for (int e = 0; e < 8; ++e) {
        // D layout: VGPR e -> M = e (lanes 0-15) / e+8 (lanes 16-31)
        const int row = rowBase + mt * 16 + e + half * 8;
        float v = acc[mt][nt][e] + bn;
        v = act ? fast_tanh(v) : fast_sigmoid(v);
        out[(size_t)row * ldo + col] = v;
      }
    }
  }
}

// ---------------------------------------------------------------------------
// f32 -> bf16 convert (weights)
// ---------------------------------------------------------------------------
__global__ void cvt_f32_bf16(const float* __restrict__ src,
                             __bf16* __restrict__ dst, int n)
{
  int i = blockIdx.x * blockDim.x + threadIdx.x;
  if (i < n) dst[i] = (__bf16)src[i];
}

// Pack combined = [input | h_cur] as bf16 [B, 2048]
__global__ void pack_combined(const float* __restrict__ x,
                              const float* __restrict__ h,
                              __bf16* __restrict__ dst)
{
  int i = blockIdx.x * blockDim.x + threadIdx.x;   // over B*2048 (exact grid)
  int row = i >> 11;
  int col = i & 2047;
  float v = (col < D_IN) ? x[(size_t)row * D_IN + col]
                         : h[(size_t)row * D_H + (col - D_IN)];
  dst[i] = (__bf16)v;
}

// ---------------------------------------------------------------------------
// Elementwise stage 1: time/excited cell update, energy, h_up -> combined
// time_out holds up_time (tanh applied) on entry, new time_cell on exit.
// exc_out holds energy_init (sigmoid applied) on entry, new excited on exit.
// ---------------------------------------------------------------------------
__global__ void ew_mid(const float* __restrict__ h_cur,
                       const float* __restrict__ excited_in,
                       const float* __restrict__ time_in,
                       const float* __restrict__ absorb,
                       const float* __restrict__ ratio,
                       float* __restrict__ time_out,
                       float* __restrict__ exc_out,
                       __bf16* __restrict__ comb)
{
  int i = blockIdx.x * blockDim.x + threadIdx.x;   // over B*D_H (exact grid)
  float up   = time_out[i];
  float t    = ratio[i] * fmaxf(time_in[i] + up, 0.0f) - 1.0f;
  float leap = (t <= 0.0f) ? 1.0f : 0.0f;
  t *= (1.0f - leap);
  time_out[i] = t;

  float ex     = excited_in[i];
  float energy = leap * ex;
  float einit  = exc_out[i];
  exc_out[i]   = ex - energy + einit * (1.0f - leap) + absorb[i];

  float hup = h_cur[i] * energy;
  int row = i >> 10, col = i & 1023;
  comb[(size_t)row * K_BIG + D_IN + col] = (__bf16)hup;
}

// Elementwise stage 2: h_next = tanh((1-hw)*h_cur + hw*h_state)
// out_h holds h_state (tanh applied) on entry, h_next on exit.
__global__ void ew_final(const float* __restrict__ h_cur,
                         const float* __restrict__ hweight,
                         float* __restrict__ out_h)
{
  int i = blockIdx.x * blockDim.x + threadIdx.x;
  float hw = hweight[i];
  out_h[i] = fast_tanh((1.0f - hw) * h_cur[i] + hw * out_h[i]);
}

// ---------------------------------------------------------------------------
extern "C" void kernel_launch(void* const* d_in, const int* in_sizes, int n_in,
                              void* d_out, int out_size, void* d_ws, size_t ws_size,
                              hipStream_t stream)
{
  (void)in_sizes; (void)n_in; (void)out_size; (void)ws_size;

  const float* x    = (const float*)d_in[0];
  const float* h    = (const float*)d_in[1];
  const float* exc  = (const float*)d_in[2];
  const float* tcel = (const float*)d_in[3];
  const float* Wt_w = (const float*)d_in[4];  const float* Wt_b = (const float*)d_in[5];
  const float* Wa_w = (const float*)d_in[6];  const float* Wa_b = (const float*)d_in[7];
  const float* Eh_w = (const float*)d_in[8];  const float* Eh_b = (const float*)d_in[9];
  const float* Hw_w = (const float*)d_in[10]; const float* Hw_b = (const float*)d_in[11];
  const float* Tr_w = (const float*)d_in[12]; const float* Tr_b = (const float*)d_in[13];
  const float* Ei_w = (const float*)d_in[14]; const float* Ei_b = (const float*)d_in[15];

  const size_t N_BD = (size_t)B_DIM * D_H;            // 4M elements
  float* out_h    = (float*)d_out;                    // h_next (staging: h_state)
  float* out_exc  = out_h + N_BD;                     // excited (staging: energy_init)
  float* out_time = out_exc + N_BD;                   // time (staging: up_time)

  // Workspace layout
  __bf16* cWt  = (__bf16*)d_ws;                       // [1024, 2048]
  __bf16* cWa  = cWt + (size_t)D_H * K_BIG;
  __bf16* cHw  = cWa + (size_t)D_H * K_BIG;
  __bf16* cEh  = cHw + (size_t)D_H * K_BIG;
  __bf16* cTr  = cEh + (size_t)D_H * K_BIG;           // [1024, 1024]
  __bf16* cEi  = cTr + (size_t)D_H * D_IN;
  __bf16* comb = cEi + (size_t)D_H * D_IN;            // [4096, 2048] bf16
  float* absorb  = (float*)(comb + (size_t)B_DIM * K_BIG);
  float* ratio   = absorb + N_BD;
  float* hweight = ratio + N_BD;

  // --- weight + activation packing (bf16) ---
  const int nW2 = D_H * K_BIG;   // 2M
  const int nW1 = D_H * D_IN;    // 1M
  cvt_f32_bf16<<<(nW2 + 255) / 256, 256, 0, stream>>>(Wt_w, cWt, nW2);
  cvt_f32_bf16<<<(nW2 + 255) / 256, 256, 0, stream>>>(Wa_w, cWa, nW2);
  cvt_f32_bf16<<<(nW2 + 255) / 256, 256, 0, stream>>>(Hw_w, cHw, nW2);
  cvt_f32_bf16<<<(nW2 + 255) / 256, 256, 0, stream>>>(Eh_w, cEh, nW2);
  cvt_f32_bf16<<<(nW1 + 255) / 256, 256, 0, stream>>>(Tr_w, cTr, nW1);
  cvt_f32_bf16<<<(nW1 + 255) / 256, 256, 0, stream>>>(Ei_w, cEi, nW1);

  const int nComb = B_DIM * K_BIG;   // 8M, exact multiple of 256
  pack_combined<<<nComb / 256, 256, 0, stream>>>(x, h, comb);

  // --- stage-1 GEMMs (all depend only on combined) ---
  dim3 blk(256);
  dim3 grd(D_H / 128, B_DIM / 128);  // (8, 32)
  gemm_bf16_wmma<<<grd, blk, 0, stream>>>(comb, K_BIG, cWt, K_BIG, Wt_b,
                                          out_time, D_H, K_BIG, /*tanh*/1);
  gemm_bf16_wmma<<<grd, blk, 0, stream>>>(comb, K_BIG, cWa, K_BIG, Wa_b,
                                          absorb, D_H, K_BIG, /*sig*/0);
  gemm_bf16_wmma<<<grd, blk, 0, stream>>>(comb, K_BIG, cHw, K_BIG, Hw_b,
                                          hweight, D_H, K_BIG, /*sig*/0);
  // Tr/Ei only see the input half of combined: K=1024, same lda.
  gemm_bf16_wmma<<<grd, blk, 0, stream>>>(comb, K_BIG, cTr, D_IN, Tr_b,
                                          ratio, D_H, D_IN, /*sig*/0);
  gemm_bf16_wmma<<<grd, blk, 0, stream>>>(comb, K_BIG, cEi, D_IN, Ei_b,
                                          out_exc, D_H, D_IN, /*sig*/0);

  // --- elementwise cell update; writes h_up into combined's h half ---
  ew_mid<<<(int)(N_BD / 256), 256, 0, stream>>>(h, exc, tcel, absorb, ratio,
                                                out_time, out_exc, comb);

  // --- Eh GEMM on [input | h_up], then final blend ---
  gemm_bf16_wmma<<<grd, blk, 0, stream>>>(comb, K_BIG, cEh, K_BIG, Eh_b,
                                          out_h, D_H, K_BIG, /*tanh*/1);
  ew_final<<<(int)(N_BD / 256), 256, 0, stream>>>(h, hweight, out_h);
}
